// MoCo_7241314861694
// MI455X (gfx1250) — compile-verified, matrix operand order checked by reading, not run on hardware
//
#include <hip/hip_runtime.h>

typedef __attribute__((ext_vector_type(16))) _Float16 v16h;
typedef __attribute__((ext_vector_type(8)))  float    v8f;

#define DIMC 128
#define KC   65536
#define BC   1024
#define LC   512
#define PC   8388608
#define INV_T (1.0f / 0.09f)

// d_out flat layout (tuple return order)
#define OUT_LOGITS 0
#define OUT_LABELS (BC * LC)                    // 524288
#define OUT_QUEUE  (BC * LC + BC)               // 525312
#define OUT_PARAMK (BC * LC + BC + DIMC * KC)   // 8913920

// ---------------------------------------------------------------------------
// Row inverse-norms for q and keys (one wave32 per 128-float row), plus labels.
// ---------------------------------------------------------------------------
__global__ void k_norms(const float* __restrict__ q, const float* __restrict__ keys,
                        float* __restrict__ qinv, float* __restrict__ kinv,
                        float* __restrict__ out_labels) {
    int gid = blockIdx.x * blockDim.x + threadIdx.x;
    if (gid < BC) out_labels[gid] = 0.0f;           // labels = zeros
    int wave = gid >> 5;                             // 2048 waves: [0,BC)=q, [BC,2BC)=keys
    int lane = threadIdx.x & 31;
    const float* src = (wave < BC) ? q : keys;
    int row = (wave < BC) ? wave : wave - BC;
    float4 v = reinterpret_cast<const float4*>(src + (size_t)row * DIMC)[lane];
    float s = v.x * v.x + v.y * v.y + v.z * v.z + v.w * v.w;
    #pragma unroll
    for (int m = 16; m >= 1; m >>= 1) s += __shfl_xor(s, m, 32);
    if (lane == 0) {
        float r = 1.0f / sqrtf(s);
        if (wave < BC) qinv[row] = r; else kinv[row] = r;
    }
}

// ---------------------------------------------------------------------------
// Fused: queueT[c][d] = queue[d][c] (for contiguous column gather), and
// out_queue = copy of queue (B columns overwritten later by k_enqueue).
// 32x32 LDS tile transpose, block (32,8).
// ---------------------------------------------------------------------------
__global__ void k_transpose(const float* __restrict__ queue, float* __restrict__ qT,
                            float* __restrict__ out_queue) {
    __shared__ float tile[32][33];
    int c0 = blockIdx.x * 32;
    int d0 = blockIdx.y * 32;
    int tx = threadIdx.x, ty = threadIdx.y;
    #pragma unroll
    for (int i = 0; i < 4; ++i) {
        int d = d0 + ty + i * 8;
        size_t idx = (size_t)d * KC + (c0 + tx);
        float v = queue[idx];
        tile[ty + i * 8][tx] = v;
        out_queue[idx] = v;                          // new_queue base copy
    }
    __syncthreads();
    #pragma unroll
    for (int i = 0; i < 4; ++i) {
        int c = c0 + ty + i * 8;
        qT[(size_t)c * DIMC + (d0 + tx)] = tile[tx][ty + i * 8];
    }
}

// ---------------------------------------------------------------------------
// Logits via WMMA. One wave32 per (batch b, 16-wide L tile).
// A = gathered queueT columns (16 x 32 f16 per step), B = qn[b] broadcast into
// all 16 columns, accumulate over DIM=128 in 4 x V_WMMA_F32_16X16X32_F16.
// D column 0 (lanes 0 and 16) carries the 16 dot products.
// ---------------------------------------------------------------------------
__global__ void k_logits(const float* __restrict__ q, const float* __restrict__ qT,
                         const float* __restrict__ qinv, const int* __restrict__ sidx,
                         float* __restrict__ out_logits) {
    int lane = threadIdx.x & 31;
    int w    = blockIdx.x * (blockDim.x >> 5) + (threadIdx.x >> 5);
    int bidx = w >> 5;                // 32 tiles of 16 per batch row (L=512)
    int tile = w & 31;
    int l0   = tile << 4;
    int m    = lane & 15;             // A-matrix row for this lane
    int base = (lane >> 4) << 3;      // K sub-offset: 0 for lanes 0-15, 8 for 16-31

    int col = sidx[bidx * LC + l0 + m];              // gathered queue column
    const float* arow = qT + (size_t)col * DIMC;     // contiguous 128 floats
    const float* qrow = q + (size_t)bidx * DIMC;
    float qs = qinv[bidx];

    v8f acc = {};
    #pragma unroll
    for (int kb = 0; kb < DIMC; kb += 32) {
        // Per ISA layout: lane holds K = base+{0..7} and base+16+{0..7}
        const float4* pa = reinterpret_cast<const float4*>(arow + kb + base);
        float4 a0 = pa[0], a1 = pa[1];
        const float4* pa2 = reinterpret_cast<const float4*>(arow + kb + base + 16);
        float4 a2 = pa2[0], a3 = pa2[1];
        const float4* pb = reinterpret_cast<const float4*>(qrow + kb + base);
        float4 b0 = pb[0], b1 = pb[1];
        const float4* pb2 = reinterpret_cast<const float4*>(qrow + kb + base + 16);
        float4 b2 = pb2[0], b3 = pb2[1];

        v16h A, Bm;
        A[0]  = (_Float16)a0.x; A[1]  = (_Float16)a0.y; A[2]  = (_Float16)a0.z; A[3]  = (_Float16)a0.w;
        A[4]  = (_Float16)a1.x; A[5]  = (_Float16)a1.y; A[6]  = (_Float16)a1.z; A[7]  = (_Float16)a1.w;
        A[8]  = (_Float16)a2.x; A[9]  = (_Float16)a2.y; A[10] = (_Float16)a2.z; A[11] = (_Float16)a2.w;
        A[12] = (_Float16)a3.x; A[13] = (_Float16)a3.y; A[14] = (_Float16)a3.z; A[15] = (_Float16)a3.w;
        Bm[0]  = (_Float16)(b0.x * qs); Bm[1]  = (_Float16)(b0.y * qs);
        Bm[2]  = (_Float16)(b0.z * qs); Bm[3]  = (_Float16)(b0.w * qs);
        Bm[4]  = (_Float16)(b1.x * qs); Bm[5]  = (_Float16)(b1.y * qs);
        Bm[6]  = (_Float16)(b1.z * qs); Bm[7]  = (_Float16)(b1.w * qs);
        Bm[8]  = (_Float16)(b2.x * qs); Bm[9]  = (_Float16)(b2.y * qs);
        Bm[10] = (_Float16)(b2.z * qs); Bm[11] = (_Float16)(b2.w * qs);
        Bm[12] = (_Float16)(b3.x * qs); Bm[13] = (_Float16)(b3.y * qs);
        Bm[14] = (_Float16)(b3.z * qs); Bm[15] = (_Float16)(b3.w * qs);

        acc = __builtin_amdgcn_wmma_f32_16x16x32_f16(
            /*neg_a=*/false, A, /*neg_b=*/false, Bm,
            /*c_mod=*/(short)0, acc, /*reuse_a=*/false, /*reuse_b=*/false);
    }

    // D layout: lane 0 VGPR r -> (M=r, N=0); lane 16 VGPR r -> (M=8+r, N=0)
    if (m == 0) {
        int rbase = base;   // 0 or 8
        #pragma unroll
        for (int r = 0; r < 8; ++r)
            out_logits[bidx * LC + l0 + rbase + r] = acc[r] * INV_T;
    }
}

// ---------------------------------------------------------------------------
// Scatter normalized keys into out_queue columns [ptr, ptr+B).
// Thread map chosen so the global writes are coalesced along the column axis.
// ---------------------------------------------------------------------------
__global__ void k_enqueue(const float* __restrict__ keys, const float* __restrict__ kinv,
                          const int* __restrict__ ptr, float* __restrict__ out_queue) {
    int t = blockIdx.x * blockDim.x + threadIdx.x;   // 131072 threads
    int d = t >> 10;
    int b = t & 1023;
    int col = ptr[0] + b;
    out_queue[(size_t)d * KC + col] = keys[b * DIMC + d] * kinv[b];
}

// ---------------------------------------------------------------------------
// new_param_k = 0.7 * param_k + 0.3 * param_q, float4-vectorized stream.
// ---------------------------------------------------------------------------
__global__ void k_momentum(const float* __restrict__ pq, const float* __restrict__ pk,
                           float* __restrict__ out) {
    int t = blockIdx.x * blockDim.x + threadIdx.x;   // P/4 threads
    float4 a = reinterpret_cast<const float4*>(pq)[t];
    float4 b = reinterpret_cast<const float4*>(pk)[t];
    float4 r;
    r.x = b.x * 0.7f + a.x * 0.3f;
    r.y = b.y * 0.7f + a.y * 0.3f;
    r.z = b.z * 0.7f + a.z * 0.3f;
    r.w = b.w * 0.7f + a.w * 0.3f;
    reinterpret_cast<float4*>(out)[t] = r;
}

extern "C" void kernel_launch(void* const* d_in, const int* in_sizes, int n_in,
                              void* d_out, int out_size, void* d_ws, size_t ws_size,
                              hipStream_t stream) {
    (void)in_sizes; (void)n_in; (void)out_size; (void)ws_size;
    const float* q     = (const float*)d_in[0];
    const float* queue = (const float*)d_in[1];
    const float* keys  = (const float*)d_in[2];
    const float* pq    = (const float*)d_in[3];
    const float* pk    = (const float*)d_in[4];
    const int*   sidx  = (const int*)d_in[5];
    const int*   ptr   = (const int*)d_in[6];
    float* out = (float*)d_out;

    // workspace: queueT (K*DIM f32 = 32MB), then qinv[B], kinv[B]
    float* qT   = (float*)d_ws;
    float* qinv = qT + (size_t)KC * DIMC;
    float* kinv = qinv + BC;

    k_norms<<<256, 256, 0, stream>>>(q, keys, qinv, kinv, out + OUT_LABELS);
    k_transpose<<<dim3(KC / 32, DIMC / 32), dim3(32, 8), 0, stream>>>(queue, qT, out + OUT_QUEUE);
    k_logits<<<(BC * (LC / 16)) / 8, 256, 0, stream>>>(q, qT, qinv, sidx, out + OUT_LOGITS);
    k_enqueue<<<(BC * DIMC) / 256, 256, 0, stream>>>(keys, kinv, ptr, out + OUT_QUEUE);
    k_momentum<<<(PC / 4) / 256, 256, 0, stream>>>(pq, pk, out + OUT_PARAMK);
}
